// CoSwinAttnBlock_71786083385548
// MI455X (gfx1250) — matrix operand order repaired and checked by reading, not compiled, not run on hardware
//
#include <hip/hip_runtime.h>

// ---------------------------------------------------------------------------
// CoSwin stereo attention block for gfx1250 (MI455X), bf16 WMMA path
// + Tensor Data Mover (TDM) staging of activation tiles into LDS.
// ---------------------------------------------------------------------------

typedef unsigned short u16;
typedef __attribute__((__ext_vector_type__(16))) __bf16 bf16x16;
typedef __attribute__((__ext_vector_type__(8)))  float  f32x8;
typedef __attribute__((__ext_vector_type__(4)))  unsigned int u32x4;
typedef __attribute__((__ext_vector_type__(8)))  int  i32x8;
typedef __attribute__((__ext_vector_type__(4)))  int  i32x4;

#define DIMC   192
#define IMG    192
#define BATCH  2
#define NWYX   24          // windows per image dimension
#define NW     576         // windows per image
#define WIN2   1152        // BATCH * NW
#define WW     64          // tokens per window
#define HEADS  6
#define HD     32
#define TOK_SIDE   73728ULL          // BATCH*IMG*IMG
#define ELEMS_SIDE 14155776ULL       // TOK_SIDE*DIMC == WIN2*WW*DIMC
#define MLPH   768

struct __align__(16) U128 { unsigned int a, b, c, d; };
union Frag16 { bf16x16 v; U128 q[2]; };

__device__ __forceinline__ u16 f2bf(float x) {
  union { float f; unsigned int u; } t; t.f = x;
  unsigned int u = t.u;
  u += 0x7fffu + ((u >> 16) & 1u);   // round-to-nearest-even
  return (u16)(u >> 16);
}

// A-fragment, 16x32 bf16 (M x K). Per ISA 7.12.2: lane = hi*16 + row;
// elements 0..7 -> K = 8*hi + 0..7 ; elements 8..15 -> K = 16 + 8*hi + 0..7
__device__ __forceinline__ Frag16 load_fragA(const u16* base, int ld, int lane) {
  Frag16 f;
  const u16* p = base + (size_t)(lane & 15) * ld + ((lane >> 4) << 3);
  f.q[0] = *(const U128*)p;
  f.q[1] = *(const U128*)(p + 16);
  return f;
}

// B-fragment, 32x16 bf16 (K x N) sourced from an n-major buffer (row n, K contiguous).
// elements 0..7 -> K = 8*hi + 0..7 ; elements 8..15 -> K = 16 + 8*hi + 0..7 of column (lane&15).
__device__ __forceinline__ Frag16 load_fragB(const u16* base, int ld, int lane) {
  Frag16 f;
  const u16* p = base + (size_t)(lane & 15) * ld + ((lane >> 4) << 3);
  f.q[0] = *(const U128*)p;
  f.q[1] = *(const U128*)(p + 16);
  return f;
}

__device__ __forceinline__ f32x8 wmma_bf16(const Frag16& a, const Frag16& b, f32x8 c) {
  return __builtin_amdgcn_wmma_f32_16x16x32_bf16(false, a.v, false, b.v,
                                                 (short)0, c, false, false);
}

// ---------------------------------------------------------------------------
// TDM: 1-D contiguous tile copy global -> LDS (D# per CDNA5 ISA §8.3/8.4).
//   group0: count=1 | lds_addr[63:32] | global_addr[120:64] | type=2[127:126]
//   group1: data_size=2B[17:16] | tensor_dim0[79:48] | tensor_dim1[111:80]=1
//           | tile_dim0[127:112] | tile_dim1[143:128]=1 | dim0_stride[207:160]
// Issued by a single wave; completion via s_wait_tensorcnt + barrier.
// This toolchain exposes the 6-arg builtin (extra i32x8 group before cpol).
// ---------------------------------------------------------------------------
__device__ __forceinline__ void tdm_load_1d(unsigned lds_byte_addr,
                                            const void* gsrc,
                                            unsigned nelem /* u16 elements, <65536 */) {
  unsigned long long ga = (unsigned long long)(size_t)gsrc;
  u32x4 g0;
  g0[0] = 1u;                                             // count=1, no gather
  g0[1] = lds_byte_addr;                                  // lds_addr
  g0[2] = (unsigned)(ga & 0xFFFFFFFFu);                   // global_addr[31:0]
  g0[3] = (unsigned)((ga >> 32) & 0x01FFFFFFu) | (0x2u << 30);  // addr hi | type=2
  i32x8 g1;
  g1[0] = (int)(1u << 16);                                // data_size = 2 bytes
  g1[1] = (int)((nelem & 0xFFFFu) << 16);                 // tensor_dim0 lo16 @bit48
  g1[2] = (int)(((nelem >> 16) & 0xFFFFu) | (1u << 16));  // tensor_dim0 hi | tensor_dim1=1
  g1[3] = (int)((nelem & 0xFFFFu) << 16);                 // tile_dim0 @bits127:112
  g1[4] = 1;                                              // tile_dim1=1, tile_dim2=0
  g1[5] = (int)nelem;                                     // tensor_dim0_stride lo32
  g1[6] = 0;
  g1[7] = 0;
  i32x4 z4 = {0, 0, 0, 0};
  i32x8 z8 = {0, 0, 0, 0, 0, 0, 0, 0};
  __builtin_amdgcn_tensor_load_to_lds(g0, g1, z4, z4, z8, 0);
}

// ---------------------------------------------------------------------------
// 1) fp32 weight -> bf16 transposed (n-major) so B-frag loads are contiguous.
// ---------------------------------------------------------------------------
__global__ __launch_bounds__(256) void wconv_kernel(const float* __restrict__ src,
                                                    u16* __restrict__ dst,
                                                    int K, int N) {
  int gid = blockIdx.x * 256 + threadIdx.x;
  if (gid >= K * N) return;
  int k = gid / N, n = gid % N;
  dst[(size_t)n * K + k] = f2bf(src[gid]);
}

// ---------------------------------------------------------------------------
// 2) LayerNorm, one wave per 192-ch token.
// ---------------------------------------------------------------------------
__global__ __launch_bounds__(256) void ln1_kernel(const float* __restrict__ x,
                                                  const float* __restrict__ g,
                                                  const float* __restrict__ bt,
                                                  float* __restrict__ out) {
  int wave = threadIdx.x >> 5, lane = threadIdx.x & 31;
  size_t tok = (size_t)blockIdx.x * 8 + wave;
  const float* xp = x + tok * DIMC;
  float v[6]; float s = 0.f;
#pragma unroll
  for (int j = 0; j < 6; j++) { v[j] = xp[lane + 32 * j]; s += v[j]; }
#pragma unroll
  for (int o = 16; o >= 1; o >>= 1) s += __shfl_xor(s, o, 32);
  float mu = s * (1.f / 192.f);
  float var = 0.f;
#pragma unroll
  for (int j = 0; j < 6; j++) { float d = v[j] - mu; var += d * d; }
#pragma unroll
  for (int o = 16; o >= 1; o >>= 1) var += __shfl_xor(var, o, 32);
  float inv = rsqrtf(var * (1.f / 192.f) + 1e-5f);
  float* op = out + tok * DIMC;
#pragma unroll
  for (int j = 0; j < 6; j++) {
    int c = lane + 32 * j;
    op[c] = (v[j] - mu) * inv * g[c] + bt[c];
  }
}

__global__ __launch_bounds__(256) void ln2_kernel(const float* __restrict__ x,
                                                  const float* __restrict__ g,
                                                  const float* __restrict__ bt,
                                                  u16* __restrict__ out) {
  int wave = threadIdx.x >> 5, lane = threadIdx.x & 31;
  size_t tok = (size_t)blockIdx.x * 8 + wave;
  const float* xp = x + tok * DIMC;
  float v[6]; float s = 0.f;
#pragma unroll
  for (int j = 0; j < 6; j++) { v[j] = xp[lane + 32 * j]; s += v[j]; }
#pragma unroll
  for (int o = 16; o >= 1; o >>= 1) s += __shfl_xor(s, o, 32);
  float mu = s * (1.f / 192.f);
  float var = 0.f;
#pragma unroll
  for (int j = 0; j < 6; j++) { float d = v[j] - mu; var += d * d; }
#pragma unroll
  for (int o = 16; o >= 1; o >>= 1) var += __shfl_xor(var, o, 32);
  float inv = rsqrtf(var * (1.f / 192.f) + 1e-5f);
  u16* op = out + tok * DIMC;
#pragma unroll
  for (int j = 0; j < 6; j++) {
    int c = lane + 32 * j;
    op[c] = f2bf((v[j] - mu) * inv * g[c] + bt[c]);
  }
}

// ---------------------------------------------------------------------------
// 3) epipolar gather + shift + window partition -> bf16 window tensors.
// ---------------------------------------------------------------------------
__global__ __launch_bounds__(256) void gather_kernel(const float* __restrict__ lnL,
                                                     const float* __restrict__ lnR,
                                                     const float* __restrict__ dL,
                                                     const float* __restrict__ dR,
                                                     u16* __restrict__ xlw,
                                                     u16* __restrict__ xrw,
                                                     u16* __restrict__ xlsw,
                                                     u16* __restrict__ xrsw) {
  size_t gid = (size_t)blockIdx.x * 256 + threadIdx.x;
  if (gid >= ELEMS_SIDE) return;
  int c  = (int)(gid % DIMC);
  int t  = (int)((gid / DIMC) % WW);
  int w2 = (int)(gid / (DIMC * WW));
  int b  = w2 / NW;
  int wy = (w2 % NW) / NWYX, wx = w2 % NWYX;
  int ty = t >> 3, tx = t & 7;
  int y = (wy * 8 + ty + 4) % IMG;      // shift roll (-4,-4)
  int x = (wx * 8 + tx + 4) % IMG;
  size_t rowbase = ((size_t)b * IMG + y) * IMG;
  size_t pix = rowbase + x;
  float dr = dR[pix], dl = dL[pix];
  int jr = (int)((float)x + 0.5f + dr); if (jr > IMG - 1) jr = IMG - 1;
  int jl = (int)fmaxf(0.f, (float)x + 0.5f - dl);
  xlw[gid]  = f2bf(lnL[pix * DIMC + c]);
  xrw[gid]  = f2bf(lnR[pix * DIMC + c]);
  xlsw[gid] = f2bf(lnL[(rowbase + jr) * DIMC + c]);   // x_left gathered at l2r
  xrsw[gid] = f2bf(lnR[(rowbase + jl) * DIMC + c]);   // x_right gathered at r2l
}

// ---------------------------------------------------------------------------
// 4) QKV GEMM per (dir, window). TDM stages both 24KB activation tiles in LDS.
//    Q,K stored [dir][win][h][tok][d] bf16;  V stored transposed [h][d][tok].
// ---------------------------------------------------------------------------
__global__ __launch_bounds__(256) void qkv_kernel(const u16* __restrict__ xlw,
                                                  const u16* __restrict__ xrw,
                                                  const u16* __restrict__ xlsw,
                                                  const u16* __restrict__ xrsw,
                                                  const u16* __restrict__ qWt,
                                                  const u16* __restrict__ kvWt,
                                                  const float* __restrict__ q_b,
                                                  const float* __restrict__ kv_b,
                                                  u16* __restrict__ Q,
                                                  u16* __restrict__ K,
                                                  u16* __restrict__ Vt) {
  __shared__ __align__(16) u16 Aq[WW * DIMC];    // 24 KB
  __shared__ __align__(16) u16 Akv[WW * DIMC];   // 24 KB
  int blk = blockIdx.x;
  int dir = blk / WIN2, win = blk % WIN2;
  const u16* Xq  = dir ? xrw  : xlw;
  const u16* Xkv = dir ? xlsw : xrsw;
  int lane = threadIdx.x & 31, wave = threadIdx.x >> 5, hi = lane >> 4;

  if (wave == 0) {   // wave-uniform branch: exactly one wave issues the DMAs
    tdm_load_1d((unsigned)(size_t)&Aq[0],  Xq  + (size_t)win * WW * DIMC, WW * DIMC);
    tdm_load_1d((unsigned)(size_t)&Akv[0], Xkv + (size_t)win * WW * DIMC, WW * DIMC);
    __builtin_amdgcn_s_wait_tensorcnt(0);
  }
  __syncthreads();

  for (int t = wave; t < 144; t += 8) {
    int kind = t / 48;               // 0=Q 1=K 2=V
    int g = t % 48;
    int mt = g & 3, nt = g >> 2;     // 4 M-tiles x 12 N-tiles
    const u16* Abase = (kind == 0 ? Aq : Akv) + (size_t)(mt * 16) * DIMC;
    const u16* Wt = (kind == 0) ? qWt : kvWt;
    int nrow0 = nt * 16 + (kind == 2 ? DIMC : 0);    // weight output row
    const u16* Bbase = Wt + (size_t)nrow0 * DIMC;
    f32x8 acc = {};
#pragma unroll
    for (int k0 = 0; k0 < DIMC; k0 += 32) {
      Frag16 a = load_fragA(Abase + k0, DIMC, lane);
      Frag16 b = load_fragB(Bbase + k0, DIMC, lane);
      acc = wmma_bf16(a, b, acc);
    }
    int cloc = nt * 16 + (lane & 15);       // 0..191
    int h = cloc >> 5, d = cloc & 31;
    size_t hb = (((size_t)dir * WIN2 + win) * HEADS + h);
#pragma unroll
    for (int e = 0; e < 8; e++) {
      int tok = mt * 16 + e + 8 * hi;
      float val = acc[e];
      if (kind == 0)       Q [(hb * WW + tok) * HD + d] = f2bf(val + q_b[cloc]);
      else if (kind == 1)  K [(hb * WW + tok) * HD + d] = f2bf(val + kv_b[cloc]);
      else                 Vt[(hb * HD + d) * WW + tok] = f2bf(val + kv_b[DIMC + cloc]);
    }
  }
}

// ---------------------------------------------------------------------------
// 5) Attention per window: both directions resident so occlusion score m can
//    be computed from both prob maps. Scores & probs in LDS.
// ---------------------------------------------------------------------------
__global__ __launch_bounds__(256) void attn_kernel(const u16* __restrict__ Q,
                                                   const u16* __restrict__ K,
                                                   const u16* __restrict__ Vt,
                                                   const float* __restrict__ rbt,
                                                   float* __restrict__ mArr,
                                                   u16* __restrict__ Obf) {
  __shared__ __align__(16) float P[2][WW][WW];
  __shared__ __align__(16) u16   Pb[2][WW][WW];
  __shared__ int reg[WW];

  int win = blockIdx.x;
  int tid = threadIdx.x, lane = tid & 31, wave = tid >> 5, hi = lane >> 4;
  int wy = (win % NW) / NWYX, wx = win % NWYX;

  if (tid < WW) {  // shift-mask region ids (Swin img_mask, un-rolled coords)
    int ty = tid >> 3, tx = tid & 7;
    int py = wy * 8 + ty, px = wx * 8 + tx;
    int rr = py < (IMG - 8) ? 0 : (py < (IMG - 4) ? 1 : 2);
    int rc = px < (IMG - 8) ? 0 : (px < (IMG - 4) ? 1 : 2);
    reg[tid] = rr * 3 + rc;
  }
  __syncthreads();

  const float scale = 0.17677669529663687f;  // 1/sqrt(32)

  for (int h = 0; h < HEADS; h++) {
    // ---- scores: 32 WMMA tiles (2 dirs x 4x4), 4 per wave ----
#pragma unroll
    for (int ti = 0; ti < 4; ti++) {
      int t = wave * 4 + ti;
      int dir = t >> 4, g = t & 15, mt = g & 3, nt = g >> 2;
      size_t hb = (((size_t)dir * WIN2 + win) * HEADS + h) * WW;
      Frag16 a = load_fragA(Q + (hb + mt * 16) * HD, HD, lane);
      Frag16 b = load_fragB(K + (hb + nt * 16) * HD, HD, lane);
      f32x8 acc = {};
      acc = wmma_bf16(a, b, acc);
      int j = nt * 16 + (lane & 15);
#pragma unroll
      for (int e = 0; e < 8; e++) {
        int i = mt * 16 + e + 8 * hi;
        int idx = (((i >> 3) - (j >> 3) + 7) * 15 + ((i & 7) - (j & 7) + 7));
        float bias = rbt[idx * HEADS + h];
        float msk = (reg[i] != reg[j]) ? -100.f : 0.f;
        P[dir][i][j] = acc[e] * scale + bias + msk;
      }
    }
    __syncthreads();

    // ---- softmax: 2 threads per row (paired lanes share via shfl_xor 1) ----
    {
      int dir = tid >> 7, r = (tid >> 1) & 63, half = tid & 1;
      float* row = P[dir][r];
      float mx = -1e30f;
#pragma unroll
      for (int j = 0; j < 32; j++) mx = fmaxf(mx, row[half * 32 + j]);
      mx = fmaxf(mx, __shfl_xor(mx, 1, 32));
      float s = 0.f;
#pragma unroll
      for (int j = 0; j < 32; j++) s += __expf(row[half * 32 + j] - mx);
      s += __shfl_xor(s, 1, 32);
      float inv = 1.f / s;
#pragma unroll
      for (int j = 0; j < 32; j++) {
        float p = __expf(row[half * 32 + j] - mx) * inv;
        row[half * 32 + j] = p;
        Pb[dir][r][half * 32 + j] = f2bf(p);
      }
    }
    __syncthreads();

    // ---- occlusion score: m[i] = sum_k relax(P_a)[i,k] * P_b[k,i] ----
    if (tid < 128) {
      int dir = tid >> 6, i = tid & 63;
      const float (*Pa)[WW] = P[dir];
      const float (*Po)[WW] = P[1 - dir];
      float acc = 0.f;
      for (int k = 0; k < WW; k++) {
        float s = Pa[i][k];
        if (i >= 1) s += Pa[i - 1][k];
        if (i >= 2) s += Pa[i - 2][k];
        if (i <= 62) s += Pa[i + 1][k];
        if (i <= 61) s += Pa[i + 2][k];
        acc += s * Po[k][i];
      }
      mArr[(((size_t)dir * WIN2 + win) * HEADS + h) * WW + i] = tanhf(5.f * acc);
    }

    // ---- AV: 16 WMMA tiles (2 dirs x 4x2), 2 per wave, K=64 in 2 steps ----
#pragma unroll
    for (int ti = 0; ti < 2; ti++) {
      int t = wave * 2 + ti;
      int dir = t >> 3, g = t & 7, mt = g & 3, nt = g >> 2;
      size_t hb = (((size_t)dir * WIN2 + win) * HEADS + h);
      f32x8 acc = {};
#pragma unroll
      for (int k0 = 0; k0 < WW; k0 += 32) {
        Frag16 a = load_fragA(&Pb[dir][mt * 16][k0], WW, lane);
        Frag16 b = load_fragB(Vt + (hb * HD + nt * 16) * WW + k0, WW, lane);
        acc = wmma_bf16(a, b, acc);
      }
      int d = nt * 16 + (lane & 15);
#pragma unroll
      for (int e = 0; e < 8; e++) {
        int tok = mt * 16 + e + 8 * hi;
        Obf[(((size_t)dir * WIN2 + win) * WW + tok) * DIMC + h * HD + d] = f2bf(acc[e]);
      }
    }
    __syncthreads();
  }
}

// ---------------------------------------------------------------------------
// 6) proj GEMM (TDM-staged A) + head-wise fuse + window reverse + residual.
// ---------------------------------------------------------------------------
__global__ __launch_bounds__(256) void proj_fuse_kernel(const u16* __restrict__ Obf,
                                                        const u16* __restrict__ projWt,
                                                        const float* __restrict__ proj_b,
                                                        const float* __restrict__ mArr,
                                                        const float* __restrict__ lnL,
                                                        const float* __restrict__ lnR,
                                                        const float* __restrict__ x_left,
                                                        const float* __restrict__ x_right,
                                                        float* __restrict__ out) {
  __shared__ __align__(16) u16 Ao[WW * DIMC];    // 24 KB
  int blk = blockIdx.x;
  int dir = blk / WIN2, win = blk % WIN2;
  int lane = threadIdx.x & 31, wave = threadIdx.x >> 5, hi = lane >> 4;
  int b = win / NW, wy = (win % NW) / NWYX, wx = win % NWYX;
  const float* lnImg = dir ? lnR : lnL;
  const float* xin   = dir ? x_right : x_left;
  float* o = out + (size_t)dir * ELEMS_SIDE;

  if (wave == 0) {
    tdm_load_1d((unsigned)(size_t)&Ao[0],
                Obf + (((size_t)dir * WIN2 + win) * WW) * DIMC, WW * DIMC);
    __builtin_amdgcn_s_wait_tensorcnt(0);
  }
  __syncthreads();

  for (int t = wave; t < 48; t += 8) {
    int mt = t & 3, nt = t >> 2;
    const u16* Ab = Ao + (size_t)(mt * 16) * DIMC;
    const u16* Bb = projWt + (size_t)(nt * 16) * DIMC;
    f32x8 acc = {};
#pragma unroll
    for (int k0 = 0; k0 < DIMC; k0 += 32) {
      Frag16 a = load_fragA(Ab + k0, DIMC, lane);
      Frag16 bfr = load_fragB(Bb + k0, DIMC, lane);
      acc = wmma_bf16(a, bfr, acc);
    }
    int c = nt * 16 + (lane & 15);
    int hH = c >> 5;
    float pb = proj_b[c];
#pragma unroll
    for (int e = 0; e < 8; e++) {
      int tok = mt * 16 + e + 8 * hi;
      int ty = tok >> 3, tx = tok & 7;
      int y = (wy * 8 + ty + 4) % IMG;     // == source of shifted window AND
      int x = (wx * 8 + tx + 4) % IMG;     //    destination after reverse roll
      size_t nidx = ((size_t)b * IMG + y) * IMG + x;
      float m  = mArr[(((size_t)dir * WIN2 + win) * HEADS + hH) * WW + tok];
      float xw = lnImg[nidx * DIMC + c];
      float xt = acc[e] + pb;
      o[nidx * DIMC + c] = xin[nidx * DIMC + c] + xw * (1.f - m) + xt * m;
    }
  }
}

// ---------------------------------------------------------------------------
// 7/8) MLP GEMMs (fc1 + exact GELU with TDM-staged A, fc2 into d_out).
// ---------------------------------------------------------------------------
__global__ __launch_bounds__(256) void fc1_kernel(const u16* __restrict__ lnBuf,
                                                  const u16* __restrict__ fc1Wt,
                                                  const float* __restrict__ fc1_b,
                                                  u16* __restrict__ H) {
  __shared__ __align__(16) u16 Aln[64 * DIMC];   // 24 KB
  int row0 = blockIdx.x * 64;
  int col0 = blockIdx.y * 64;
  int lane = threadIdx.x & 31, wave = threadIdx.x >> 5, hi = lane >> 4;

  if (wave == 0) {
    tdm_load_1d((unsigned)(size_t)&Aln[0], lnBuf + (size_t)row0 * DIMC, 64 * DIMC);
    __builtin_amdgcn_s_wait_tensorcnt(0);
  }
  __syncthreads();

#pragma unroll
  for (int ti = 0; ti < 2; ti++) {
    int t = wave * 2 + ti;
    int mt = t & 3, nt = t >> 2;
    const u16* Ab = Aln + (size_t)(mt * 16) * DIMC;
    const u16* Bb = fc1Wt + ((size_t)(col0 + nt * 16)) * DIMC;
    f32x8 acc = {};
#pragma unroll
    for (int k0 = 0; k0 < DIMC; k0 += 32) {
      Frag16 a = load_fragA(Ab + k0, DIMC, lane);
      Frag16 b = load_fragB(Bb + k0, DIMC, lane);
      acc = wmma_bf16(a, b, acc);
    }
    int c = col0 + nt * 16 + (lane & 15);
    float bias = fc1_b[c];
#pragma unroll
    for (int e = 0; e < 8; e++) {
      int row = row0 + mt * 16 + e + 8 * hi;
      float v = acc[e] + bias;
      float gv = 0.5f * v * (1.f + erff(v * 0.70710678118654752f));  // exact gelu
      H[(size_t)row * MLPH + c] = f2bf(gv);
    }
  }
}

__global__ __launch_bounds__(256) void fc2_kernel(const u16* __restrict__ H,
                                                  const u16* __restrict__ fc2Wt,
                                                  const float* __restrict__ fc2_b,
                                                  float* __restrict__ out) {
  int row0 = blockIdx.x * 64;
  int col0 = blockIdx.y * 64;
  int lane = threadIdx.x & 31, wave = threadIdx.x >> 5, hi = lane >> 4;
#pragma unroll
  for (int ti = 0; ti < 2; ti++) {
    int t = wave * 2 + ti;
    int mt = t & 3, nt = t >> 2;
    const u16* Ab = H + ((size_t)(row0 + mt * 16)) * MLPH;
    const u16* Bb = fc2Wt + ((size_t)(col0 + nt * 16)) * MLPH;
    f32x8 acc = {};
    for (int k0 = 0; k0 < MLPH; k0 += 32) {
      Frag16 a = load_fragA(Ab + k0, MLPH, lane);
      Frag16 b = load_fragB(Bb + k0, MLPH, lane);
      acc = wmma_bf16(a, b, acc);
    }
    int c = col0 + nt * 16 + (lane & 15);
    float bias = fc2_b[c];
#pragma unroll
    for (int e = 0; e < 8; e++) {
      int row = row0 + mt * 16 + e + 8 * hi;
      out[(size_t)row * DIMC + c] += acc[e] + bias;   // residual add
    }
  }
}

// ---------------------------------------------------------------------------
extern "C" void kernel_launch(void* const* d_in, const int* in_sizes, int n_in,
                              void* d_out, int out_size, void* d_ws, size_t ws_size,
                              hipStream_t stream) {
  (void)in_sizes; (void)n_in; (void)out_size; (void)ws_size;
  const float* x_left   = (const float*)d_in[0];
  const float* x_right  = (const float*)d_in[1];
  const float* d_left   = (const float*)d_in[2];
  const float* d_right  = (const float*)d_in[3];
  const float* norm1_g  = (const float*)d_in[4];
  const float* norm1_b  = (const float*)d_in[5];
  const float* q_w      = (const float*)d_in[6];
  const float* q_b      = (const float*)d_in[7];
  const float* kv_w     = (const float*)d_in[8];
  const float* kv_b     = (const float*)d_in[9];
  const float* proj_w   = (const float*)d_in[10];
  const float* proj_b   = (const float*)d_in[11];
  const float* rel_bias = (const float*)d_in[12];
  const float* norm2_g  = (const float*)d_in[13];
  const float* norm2_b  = (const float*)d_in[14];
  const float* fc1_w    = (const float*)d_in[15];
  const float* fc1_b    = (const float*)d_in[16];
  const float* fc2_w    = (const float*)d_in[17];
  const float* fc2_b    = (const float*)d_in[18];

  char* wsb = (char*)d_ws;
  size_t off = 0;
  auto take = [&](size_t bytes) -> char* {
    char* p = wsb + off;
    off += (bytes + 255) & ~(size_t)255;
    return p;
  };

  float* lnL  = (float*)take(ELEMS_SIDE * 4);
  float* lnR  = (float*)take(ELEMS_SIDE * 4);
  u16* xlw    = (u16*)take(ELEMS_SIDE * 2);   // } phase2 alias: lnBuf
  u16* xrw    = (u16*)take(ELEMS_SIDE * 2);   // }
  u16* xlsw   = (u16*)take(ELEMS_SIDE * 2);
  u16* xrsw   = (u16*)take(ELEMS_SIDE * 2);
  u16* Qb     = (u16*)take(2 * ELEMS_SIDE * 2);  // } phase2 alias: Hbuf
  u16* Kb     = (u16*)take(2 * ELEMS_SIDE * 2);  // }
  u16* Vtb    = (u16*)take(2 * ELEMS_SIDE * 2);  // }
  u16* Obf    = (u16*)take(2 * ELEMS_SIDE * 2);  // }
  float* mArr = (float*)take(2ULL * WIN2 * HEADS * WW * 4);
  u16* qWt    = (u16*)take(192 * 192 * 2);
  u16* kvWt   = (u16*)take(192 * 384 * 2);
  u16* projWt = (u16*)take(192 * 192 * 2);
  u16* fc1Wt  = (u16*)take(192 * 768 * 2);
  u16* fc2Wt  = (u16*)take(768 * 192 * 2);

  u16* lnBuf = xlw;   // phase-2 overlay (56.6 MB over xlw+xrw, dead by then)
  u16* Hbuf  = Qb;    // phase-2 overlay (226.5 MB over Q/K/Vt/Obf, dead by then)

  // weights -> bf16 transposed (n-major)
  wconv_kernel<<<144, 256, 0, stream>>>(q_w,   qWt,   192, 192);
  wconv_kernel<<<288, 256, 0, stream>>>(kv_w,  kvWt,  192, 384);
  wconv_kernel<<<144, 256, 0, stream>>>(proj_w, projWt, 192, 192);
  wconv_kernel<<<576, 256, 0, stream>>>(fc1_w, fc1Wt, 192, 768);
  wconv_kernel<<<576, 256, 0, stream>>>(fc2_w, fc2Wt, 768, 192);

  // LN1 both eyes
  ln1_kernel<<<(unsigned)(TOK_SIDE / 8), 256, 0, stream>>>(x_left,  norm1_g, norm1_b, lnL);
  ln1_kernel<<<(unsigned)(TOK_SIDE / 8), 256, 0, stream>>>(x_right, norm1_g, norm1_b, lnR);

  // epipolar gather + shift + window partition
  gather_kernel<<<(unsigned)(ELEMS_SIDE / 256), 256, 0, stream>>>(
      lnL, lnR, d_left, d_right, xlw, xrw, xlsw, xrsw);

  // QKV per (dir, window), TDM-staged activations
  qkv_kernel<<<2 * WIN2, 256, 0, stream>>>(xlw, xrw, xlsw, xrsw,
                                           qWt, kvWt, q_b, kv_b, Qb, Kb, Vtb);

  // attention (both directions per window, occlusion score m)
  attn_kernel<<<WIN2, 256, 0, stream>>>(Qb, Kb, Vtb, rel_bias, mArr, Obf);

  // proj + fuse + window reverse + residual -> d_out
  proj_fuse_kernel<<<2 * WIN2, 256, 0, stream>>>(Obf, projWt, proj_b, mArr,
                                                 lnL, lnR, x_left, x_right,
                                                 (float*)d_out);

  // MLP: ln2 -> fc1(gelu) -> fc2 (+= residual into d_out)
  ln2_kernel<<<(unsigned)(2 * TOK_SIDE / 8), 256, 0, stream>>>(
      (const float*)d_out, norm2_g, norm2_b, lnBuf);
  fc1_kernel<<<dim3((unsigned)(2 * TOK_SIDE / 64), MLPH / 64), 256, 0, stream>>>(
      lnBuf, fc1Wt, fc1_b, Hbuf);
  fc2_kernel<<<dim3((unsigned)(2 * TOK_SIDE / 64), DIMC / 64), 256, 0, stream>>>(
      Hbuf, fc2Wt, fc2_b, (float*)d_out);
}
